// FPNSemanticHeadDPC_43035572306076
// MI455X (gfx1250) — compile-verified, matrix operand order checked by reading, not run on hardware
//
#include <hip/hip_runtime.h>
#include <hip/hip_bf16.h>

typedef __attribute__((ext_vector_type(16))) __bf16 v16bf;
typedef __attribute__((ext_vector_type(8)))  float  v8f;
typedef unsigned short u16;
typedef unsigned int   u32;

static constexpr int B_ = 2, C_ = 128, H_ = 64, W_ = 512;
static constexpr int HW = H_ * W_;            // 32768
static constexpr int Hp = H_ + 2, Wp = W_ + 2;
static constexpr int HpWp = Hp * Wp;          // 33924
#define ABN_EPS 1e-5f
#define ABN_SLOPE 0.01f

__device__ __forceinline__ u16 f2bf(float f) {
  u32 u = __float_as_uint(f);
  u32 r = 0x7FFFu + ((u >> 16) & 1u);   // round-to-nearest-even
  return (u16)((u + r) >> 16);
}

// ---------------------------------------------------------------------------
// bf16 GEMM via WMMA:  Z[b, co, p] = sum_k Wt[co,k] * X[b,k,p]
// X: bf16 [B][Kpad][HW], Wt: bf16 [Co][Kpad], Z: f32 [B][Co][HW]
// Block: 256 threads (8 waves). Tile M=64, N=128, K-step 32.
// ---------------------------------------------------------------------------
__global__ __launch_bounds__(256)
void gemm_bf16_k(const u16* __restrict__ X, const u16* __restrict__ Wt,
                 float* __restrict__ Z, int Kpad, int Co) {
  __shared__ __align__(16) u16 lds_x[128 * 48]; // [p 0..127][k 0..31], stride 48
  const int t = threadIdx.x;
  const int lane = t & 31, wave = t >> 5;
  const int mwave = wave & 3, nwave = wave >> 2;
  const int b  = blockIdx.z;
  const int p0 = blockIdx.x * 128;
  const int m0 = blockIdx.y * 64;
  const int halfsel = lane >> 4;            // 0: K{0..7,16..23}, 1: K{8..15,24..31}
  const int kOff = halfsel * 8;
  const int lan = lane & 15;

  v8f acc[4] = {};
  const size_t xbase = (size_t)b * Kpad * HW;
  const u16* wrow_base = Wt + (size_t)(m0 + mwave * 16 + lan) * Kpad;

  for (int k0 = 0; k0 < Kpad; k0 += 32) {
    // Cooperative load of X tile [32 k][128 p] -> LDS transposed [p][k]
#pragma unroll
    for (int i = 0; i < 2; ++i) {
      int id = t + 256 * i;                 // 0..511
      int kk = id >> 4;                     // 0..31
      int pc = id & 15;                     // 0..15 (8-element column chunks)
      const u16* gp = X + xbase + (size_t)(k0 + kk) * HW + p0 + pc * 8;
      uint4 v = *(const uint4*)gp;          // 8 bf16, coalesced along p
      const u16* s = (const u16*)&v;
      int col0 = pc * 8;
#pragma unroll
      for (int j = 0; j < 8; ++j) lds_x[(col0 + j) * 48 + kk] = s[j];
    }
    if (k0 + 32 < Kpad)
      __builtin_prefetch(wrow_base + k0 + 32, 0, 0);  // global_prefetch_b8
    __syncthreads();

    // A fragment (16x32 weights) per CDNA5 16-bit A layout
    union { uint4 q[2]; v16bf v; } afr;
    {
      const u16* wrow = wrow_base + k0;
      afr.q[0] = *(const uint4*)(wrow + kOff);        // K kOff..kOff+7
      afr.q[1] = *(const uint4*)(wrow + 16 + kOff);   // K 16+kOff..
    }

#pragma unroll
    for (int nt = 0; nt < 4; ++nt) {
      union { uint4 q[2]; v16bf v; } bfr;
      const u16* ldsc = &lds_x[(nwave * 64 + nt * 16 + lan) * 48];
      bfr.q[0] = *(const uint4*)(ldsc + kOff);
      bfr.q[1] = *(const uint4*)(ldsc + 16 + kOff);
      acc[nt] = __builtin_amdgcn_wmma_f32_16x16x32_bf16(
          false, afr.v, false, bfr.v, (short)0, acc[nt], false, false);
    }
    __syncthreads();
  }

  // Store per CDNA5 16x16 f32 C/D layout
#pragma unroll
  for (int nt = 0; nt < 4; ++nt) {
    int col = p0 + nwave * 64 + nt * 16 + lan;
#pragma unroll
    for (int r = 0; r < 8; ++r) {
      int row = m0 + mwave * 16 + r + halfsel * 8;
      Z[(size_t)(b * Co + row) * HW + col] = acc[nt][r];
    }
  }
}

// ---------------------------------------------------------------------------
// Depthwise 3x3 dilated conv (same padding), f32 in -> bf16 out
// ---------------------------------------------------------------------------
__global__ void dwconv3x3_k(const float* __restrict__ in, const float* __restrict__ wt,
                            u16* __restrict__ out, int dh, int dw, int n_total) {
  int n = blockIdx.x * blockDim.x + threadIdx.x;
  if (n >= n_total) return;
  int p = n % HW;
  int c = (n / HW) % C_;
  int b = n / (HW * C_);
  int h = p / W_, w = p % W_;
  const float* src = in + (size_t)(b * C_ + c) * HW;
  const float* w9 = wt + c * 9;
  float s = 0.f;
#pragma unroll
  for (int ky = 0; ky < 3; ++ky) {
    int ih = h + (ky - 1) * dh;
    if (ih < 0 || ih >= H_) continue;
#pragma unroll
    for (int kx = 0; kx < 3; ++kx) {
      int iw = w + (kx - 1) * dw;
      if (iw < 0 || iw >= W_) continue;
      s += src[ih * W_ + iw] * w9[ky * 3 + kx];
    }
  }
  out[n] = f2bf(s);
}

// ---------------------------------------------------------------------------
// offset = 3*lambda*tanh(1/x_range); keep f32, mirror bf16 into both concat bufs
// ---------------------------------------------------------------------------
__global__ void offset_k(const float* __restrict__ xr, const float* __restrict__ lam,
                         float* __restrict__ offs, u16* __restrict__ cpo,
                         u16* __restrict__ cpi, int n_total) {
  int n = blockIdx.x * blockDim.x + threadIdx.x;
  if (n >= n_total) return;
  float o = 3.0f * lam[0] * tanhf(1.0f / xr[n]);
  offs[n] = o;
  int p = n % HW;
  int j = (n / HW) % 9;
  int b = n / (9 * HW);
  u16 bv = f2bf(o);
  size_t di = ((size_t)b * 160 + 128 + j) * HW + p;
  cpo[di] = bv;
  cpi[di] = bv;
}

// ---------------------------------------------------------------------------
// Deformable gather + per-channel 9-tap weighted sum -> bf16 concat channels
// Faithful to source: pre_v uses W (not Wp); offset indexed as flat (p*9+k)
// over the [9*HW] array; round-half-even; clip to padded grid.
// ---------------------------------------------------------------------------
__global__ void gather_ws_k(const float* __restrict__ src, const float* __restrict__ offs,
                            const float* __restrict__ w9, u16* __restrict__ dst,
                            int dstKc, float padval, int n_total) {
  int n = blockIdx.x * blockDim.x + threadIdx.x;
  if (n >= n_total) return;
  int p = n % HW;
  int b = n / HW;
  int h = p / W_, w = p % W_;
  const int xo[9] = {-1, 0, 1, -1, 0, 1, -1, 0, 1};
  const int yo[9] = {-1, -1, -1, 0, 0, 0, 1, 1, 1};
  int lin[9];
  bool inb[9];
  const float* ob = offs + (size_t)b * 9 * HW;
#pragma unroll
  for (int k = 0; k < 9; ++k) {
    int pre = (h + 1) * Wp + (w + 1) + xo[k] + yo[k] * W_ + 2;
    float a = rintf((float)pre + ob[p * 9 + k]);
    a = fminf(fmaxf(a, 0.f), (float)(HpWp - 1));
    int idx = (int)a;
    int hp = idx / Wp, wpp = idx - hp * Wp;
    inb[k] = (hp >= 1 && hp <= H_ && wpp >= 1 && wpp <= W_);
    lin[k] = (hp - 1) * W_ + (wpp - 1);
  }
  for (int c = 0; c < C_; ++c) {
    const float* sb = src + (size_t)(b * C_ + c) * HW;
    const float* wc = w9 + c * 9;
    float s = 0.f;
#pragma unroll
    for (int k = 0; k < 9; ++k) {
      float v = inb[k] ? sb[lin[k]] : padval;
      s += v * wc[k];
    }
    dst[((size_t)b * dstKc + c) * HW + p] = f2bf(s);
  }
}

// ---------------------------------------------------------------------------
// Per-channel BN stats over (B, HW): stats[c]=mean, stats[Co+c]=rsqrt(var+eps)
// ---------------------------------------------------------------------------
__global__ __launch_bounds__(256)
void bn_stats_k(const float* __restrict__ Z, float* __restrict__ stats, int Co) {
  __shared__ float ss[256], sq[256];
  int co = blockIdx.x, t = threadIdx.x;
  float s = 0.f, q = 0.f;
  for (int b = 0; b < B_; ++b) {
    const float* zp = Z + (size_t)(b * Co + co) * HW;
    for (int p = t; p < HW; p += 256) { float v = zp[p]; s += v; q += v * v; }
  }
  ss[t] = s; sq[t] = q;
  __syncthreads();
  for (int st = 128; st > 0; st >>= 1) {
    if (t < st) { ss[t] += ss[t + st]; sq[t] += sq[t + st]; }
    __syncthreads();
  }
  if (t == 0) {
    float inv = 1.0f / (float)(B_ * HW);
    float m = ss[0] * inv;
    float var = sq[0] * inv - m * m;   // population variance (ddof=0)
    stats[co] = m;
    stats[Co + co] = rsqrtf(var + ABN_EPS);
  }
}

// ---------------------------------------------------------------------------
// ABN apply (normalize + affine + leaky relu); optional f32 copy and/or bf16
// write into a channel slice of a concat buffer.
// ---------------------------------------------------------------------------
__global__ void abn_k(const float* __restrict__ Z, const float* __restrict__ stats,
                      const float* __restrict__ g, const float* __restrict__ be,
                      float* __restrict__ outF, u16* __restrict__ outB,
                      int Co, int catKc, int catOff, int n_total) {
  int n = blockIdx.x * blockDim.x + threadIdx.x;
  if (n >= n_total) return;
  int p = n % HW;
  int c = (n / HW) % Co;
  int b = n / (HW * Co);
  float y = (Z[n] - stats[c]) * stats[Co + c] * g[c] + be[c];
  y = (y >= 0.f) ? y : ABN_SLOPE * y;
  if (outF) outF[n] = y;
  if (outB) outB[((size_t)b * catKc + catOff + c) * HW + p] = f2bf(y);
}

// Zero the K-pad channels [137,160) of both (C+9)-concat bf16 buffers
__global__ void zero_pad_k(u16* __restrict__ a, u16* __restrict__ c2, int n_total) {
  int n = blockIdx.x * blockDim.x + threadIdx.x;
  if (n >= n_total) return;
  int p = n % HW;
  int cr = (n / HW) % 23;
  int b = n / (23 * HW);
  size_t di = ((size_t)b * 160 + 137 + cr) * HW + p;
  a[di] = 0;
  c2[di] = 0;
}

// Convert f32 weight [M][Ks] -> bf16 [M][Kd] with zero K-padding
__global__ void wconv_k(const float* __restrict__ src, u16* __restrict__ dst,
                        int M, int Ks, int Kd) {
  int n = blockIdx.x * blockDim.x + threadIdx.x;
  if (n >= M * Kd) return;
  int m = n / Kd, k = n % Kd;
  dst[n] = (k < Ks) ? f2bf(src[(size_t)m * Ks + k]) : (u16)0;
}

// ---------------------------------------------------------------------------
extern "C" void kernel_launch(void* const* d_in, const int* in_sizes, int n_in,
                              void* d_out, int out_size, void* d_ws, size_t ws_size,
                              hipStream_t stream) {
  const float* x      = (const float*)d_in[0];
  const float* xrange = (const float*)d_in[1];
  const float* dw1 = (const float*)d_in[2];
  const float* pw1 = (const float*)d_in[3];
  const float* g1  = (const float*)d_in[4];
  const float* b1  = (const float*)d_in[5];
  const float* dw2 = (const float*)d_in[6];
  const float* pw2 = (const float*)d_in[7];
  const float* g2  = (const float*)d_in[8];
  const float* b2  = (const float*)d_in[9];
  const float* dw3 = (const float*)d_in[10];
  const float* pw3 = (const float*)d_in[11];
  const float* g3  = (const float*)d_in[12];
  const float* b3  = (const float*)d_in[13];
  const float* dw4 = (const float*)d_in[14];
  const float* pw4 = (const float*)d_in[15];
  const float* g4  = (const float*)d_in[16];
  const float* b4  = (const float*)d_in[17];
  const float* range_in  = (const float*)d_in[18];
  const float* range_out = (const float*)d_in[19];
  const float* lam  = (const float*)d_in[20];
  const float* w_ro = (const float*)d_in[21];
  const float* g_ro = (const float*)d_in[22];
  const float* b_ro = (const float*)d_in[23];
  const float* w_ri = (const float*)d_in[24];
  const float* g_ri = (const float*)d_in[25];
  const float* b_ri = (const float*)d_in[26];
  const float* w2   = (const float*)d_in[27];
  const float* g_out = (const float*)d_in[28];
  const float* b_out = (const float*)d_in[29];

  // Workspace carve-out (256B aligned slices)
  char* ws = (char*)d_ws;
  size_t off = 0;
  auto alloc = [&](size_t bytes) -> void* {
    void* p = ws + off;
    off += (bytes + 255) & ~(size_t)255;
    return p;
  };
  float* offs  = (float*)alloc((size_t)B_ * 9 * HW * 4);    // f32 offsets
  u16* cat_po  = (u16*)alloc((size_t)B_ * 160 * HW * 2);    // [x_po | offset | pad]
  u16* cat_pi  = (u16*)alloc((size_t)B_ * 160 * HW * 2);    // [x_pi | offset | pad]
  u16* buf_dw  = (u16*)alloc((size_t)B_ * 128 * HW * 2);    // depthwise out (bf16)
  float* zb    = (float*)alloc((size_t)B_ * 128 * HW * 4);  // GEMM out (f32)
  float* xa    = (float*)alloc((size_t)B_ * 128 * HW * 4);  // abn(sep_conv1) f32
  float* x3f   = (float*)alloc((size_t)B_ * 128 * HW * 4);  // x3 f32
  u16* cat     = (u16*)alloc((size_t)B_ * 768 * HW * 2);    // 6C concat (bf16)
  u16* pw1b = (u16*)alloc(128 * 128 * 2);
  u16* pw2b = (u16*)alloc(128 * 128 * 2);
  u16* pw3b = (u16*)alloc(128 * 128 * 2);
  u16* pw4b = (u16*)alloc(128 * 128 * 2);
  u16* wrob = (u16*)alloc(64 * 160 * 2);
  u16* wrib = (u16*)alloc(64 * 160 * 2);
  u16* w2b  = (u16*)alloc(128 * 768 * 2);
  float* stats = (float*)alloc(256 * 4);

  const int TB = 256;
  auto nb = [](long n) { return (unsigned)((n + 255) / 256); };
  const int nPix = B_ * HW;             // 65536
  const int nC   = B_ * 128 * HW;       // 8.39M
  const int nH   = B_ * 64 * HW;

  // Weight conversion (bf16, K-padded)
  wconv_k<<<nb(128 * 128), TB, 0, stream>>>(pw1, pw1b, 128, 128, 128);
  wconv_k<<<nb(128 * 128), TB, 0, stream>>>(pw2, pw2b, 128, 128, 128);
  wconv_k<<<nb(128 * 128), TB, 0, stream>>>(pw3, pw3b, 128, 128, 128);
  wconv_k<<<nb(128 * 128), TB, 0, stream>>>(pw4, pw4b, 128, 128, 128);
  wconv_k<<<nb(64 * 160), TB, 0, stream>>>(w_ro, wrob, 64, 137, 160);
  wconv_k<<<nb(64 * 160), TB, 0, stream>>>(w_ri, wrib, 64, 137, 160);
  wconv_k<<<nb(128 * 768), TB, 0, stream>>>(w2, w2b, 128, 768, 768);
  zero_pad_k<<<nb(B_ * 23 * HW), TB, 0, stream>>>(cat_po, cat_pi, B_ * 23 * HW);

  // Offsets (f32 + bf16 mirrors into both concat inputs)
  offset_k<<<nb(B_ * 9 * HW), TB, 0, stream>>>(xrange, lam, offs, cat_po, cat_pi,
                                               B_ * 9 * HW);

  // x_po = abn(conv1x1([gather_ws(x,0.5,range_out) | offset], w_ro)) -> cat[640:704)
  gather_ws_k<<<nb(nPix), TB, 0, stream>>>(x, offs, range_out, cat_po, 160, 0.5f, nPix);
  gemm_bf16_k<<<dim3(HW / 128, 1, B_), TB, 0, stream>>>(cat_po, wrob, zb, 160, 64);
  bn_stats_k<<<64, TB, 0, stream>>>(zb, stats, 64);
  abn_k<<<nb(nH), TB, 0, stream>>>(zb, stats, g_ro, b_ro, nullptr, cat, 64, 768, 640, nH);

  // xa = abn(sep_conv(x, dw1, pw1, (1,6))) -> f32 + cat[0:128)
  dwconv3x3_k<<<nb(nC), TB, 0, stream>>>(x, dw1, buf_dw, 1, 6, nC);
  gemm_bf16_k<<<dim3(HW / 128, 2, B_), TB, 0, stream>>>(buf_dw, pw1b, zb, 128, 128);
  bn_stats_k<<<128, TB, 0, stream>>>(zb, stats, 128);
  abn_k<<<nb(nC), TB, 0, stream>>>(zb, stats, g1, b1, xa, cat, 128, 768, 0, nC);

  // x_pi = abn(conv1x1([gather_ws(xa,0.0,range_in) | offset], w_ri)) -> cat[704:768)
  gather_ws_k<<<nb(nPix), TB, 0, stream>>>(xa, offs, range_in, cat_pi, 160, 0.0f, nPix);
  gemm_bf16_k<<<dim3(HW / 128, 1, B_), TB, 0, stream>>>(cat_pi, wrib, zb, 160, 64);
  bn_stats_k<<<64, TB, 0, stream>>>(zb, stats, 64);
  abn_k<<<nb(nH), TB, 0, stream>>>(zb, stats, g_ri, b_ri, nullptr, cat, 64, 768, 704, nH);

  // x1 = abn(sep_conv(xa, dw2, pw2, (1,1))) -> cat[128:256)
  dwconv3x3_k<<<nb(nC), TB, 0, stream>>>(xa, dw2, buf_dw, 1, 1, nC);
  gemm_bf16_k<<<dim3(HW / 128, 2, B_), TB, 0, stream>>>(buf_dw, pw2b, zb, 128, 128);
  bn_stats_k<<<128, TB, 0, stream>>>(zb, stats, 128);
  abn_k<<<nb(nC), TB, 0, stream>>>(zb, stats, g2, b2, nullptr, cat, 128, 768, 128, nC);

  // x2 = abn(sep_conv(xa, dw3, pw3, (6,21))) -> cat[256:384)
  dwconv3x3_k<<<nb(nC), TB, 0, stream>>>(xa, dw3, buf_dw, 6, 21, nC);
  gemm_bf16_k<<<dim3(HW / 128, 2, B_), TB, 0, stream>>>(buf_dw, pw3b, zb, 128, 128);
  bn_stats_k<<<128, TB, 0, stream>>>(zb, stats, 128);
  abn_k<<<nb(nC), TB, 0, stream>>>(zb, stats, g3, b3, nullptr, cat, 128, 768, 256, nC);

  // x3 = abn(sep_conv(xa, dw4, pw4, (18,15))) -> f32 + cat[384:512)
  dwconv3x3_k<<<nb(nC), TB, 0, stream>>>(xa, dw4, buf_dw, 18, 15, nC);
  gemm_bf16_k<<<dim3(HW / 128, 2, B_), TB, 0, stream>>>(buf_dw, pw4b, zb, 128, 128);
  bn_stats_k<<<128, TB, 0, stream>>>(zb, stats, 128);
  abn_k<<<nb(nC), TB, 0, stream>>>(zb, stats, g4, b4, x3f, cat, 128, 768, 384, nC);

  // x4 = abn(sep_conv(x3, dw4, pw4, (18,15))) (conv4 reused) -> cat[512:640)
  dwconv3x3_k<<<nb(nC), TB, 0, stream>>>(x3f, dw4, buf_dw, 18, 15, nC);
  gemm_bf16_k<<<dim3(HW / 128, 2, B_), TB, 0, stream>>>(buf_dw, pw4b, zb, 128, 128);
  bn_stats_k<<<128, TB, 0, stream>>>(zb, stats, 128);
  abn_k<<<nb(nC), TB, 0, stream>>>(zb, stats, g4, b4, nullptr, cat, 128, 768, 512, nC);

  // out = abn(conv1x1(cat, w2))
  gemm_bf16_k<<<dim3(HW / 128, 2, B_), TB, 0, stream>>>(cat, w2b, zb, 768, 128);
  bn_stats_k<<<128, TB, 0, stream>>>(zb, stats, 128);
  abn_k<<<nb(nC), TB, 0, stream>>>(zb, stats, g_out, b_out, (float*)d_out, nullptr,
                                   128, 0, 0, nC);
}
